// EfficientCrossAttention_66236985639309
// MI455X (gfx1250) — matrix operand order repaired and checked by reading, not compiled, not run on hardware
//
#include <hip/hip_runtime.h>
#include <hip/hip_bf16.h>

// ---------------- problem constants (from reference) ----------------
#define EMBED   256
#define HEADS   8
#define HD      32          // head dim
#define NQ      4096        // H_FEAT*W_FEAT
#define NSP     16384       // spatial tokens before pooling (4 frames * 64*64)
#define NSPP    4096        // pooled spatial tokens (4 * 32*32)
#define NKV     4160        // pooled kv tokens (NSPP + 64 pointer)
#define SCALE_F 0.17677669529663687f   // 32^-0.5
#define POOLC_F 1.3862943611198906f    // ln(4)
#define LOG2E   1.4426950408889634f
// exp(x) computed as exp2(x*log2e); fold scale + pool bias - safety offset (8)
#define XSCALE  (SCALE_F * LOG2E)
#define BADD_SP ((POOLC_F - 8.0f) * LOG2E)
#define BADD_PT ((0.0f    - 8.0f) * LOG2E)

typedef __attribute__((ext_vector_type(16))) _Float16 v16h;
typedef __attribute__((ext_vector_type(8)))  float    v8f;

// raw v_exp_f32 / v_rcp_f32 (arguments here are far from the denormal-fixup range)
__device__ __forceinline__ float fast_exp2(float x) { return __builtin_amdgcn_exp2f(x); }
__device__ __forceinline__ float fast_rcp(float x)  { return __builtin_amdgcn_rcpf(x); }

// ---------------- WMMA wrapper ----------------
__device__ __forceinline__ v8f wmma_f16(v16h a, v16h b, v8f c) {
    return __builtin_amdgcn_wmma_f32_16x16x32_f16(
        false, a, false, b, (short)0, c, false, false);
}

// ---------------- fragment loaders (wave32; this code's contiguous k order, same for A and B) ----------------
// A 16x32 f16: lane L -> row M=L&15, half=L>>4; VGPR pair v: K = 16*half + 2v
__device__ __forceinline__ v16h load_A_tile(const _Float16* A, int ld, int r0, int c0, int lane) {
    int row = lane & 15, half = lane >> 4;
    const _Float16* p = A + (size_t)(r0 + row) * ld + c0;
    v16h a;
#pragma unroll
    for (int v = 0; v < 8; ++v) {
        int k = half * 16 + 2 * v;
        a[2 * v]     = p[k];
        a[2 * v + 1] = p[k + 1];
    }
    return a;
}

// B 32x16 f16 where the source X is row-major [n][k] (i.e. B = X^T):
// lane L -> col N=L&15, half=L>>4; VGPR pair v: K = 16*half + 2v (contiguous in X row)
__device__ __forceinline__ v16h load_B_transposed(const _Float16* X, int ld, int n0, int c0, int lane) {
    int n = lane & 15, half = lane >> 4;
    const _Float16* p = X + (size_t)(n0 + n) * ld + c0;
    v16h b;
#pragma unroll
    for (int v = 0; v < 8; ++v) {
        int k = half * 16 + 2 * v;
        b[2 * v]     = p[k];
        b[2 * v + 1] = p[k + 1];
    }
    return b;
}

// ---------------- kernel 0a: weight transpose + f32 -> f16 (Wt[n][k] = W[k][n]) ----------------
__global__ void transpose_w_kernel(const float* __restrict__ W, _Float16* __restrict__ Wt) {
    int k = blockIdx.x;     // row of W (coalesced read)
    int n = threadIdx.x;
    Wt[(size_t)n * EMBED + k] = (_Float16)W[(size_t)k * EMBED + n];
}

// ---------------- kernel 0b: plain f32 -> f16 convert (for q) ----------------
__global__ void cvt_f16_kernel(const float* __restrict__ in, _Float16* __restrict__ out) {
    size_t i = (size_t)blockIdx.x * EMBED + threadIdx.x;
    out[i] = (_Float16)in[i];
}

// ---------------- kernel 1: 2x2 avg pool (spatial) + pointer copy, fp32 -> f16 ----------------
__global__ void pool_kernel(const float* __restrict__ k_in, const float* __restrict__ v_in,
                            _Float16* __restrict__ kpool, _Float16* __restrict__ vpool) {
    int t = blockIdx.x;       // pooled token 0..NKV-1
    int c = threadIdx.x;      // channel 0..255
    size_t dst = (size_t)t * EMBED + c;
    if (t < NSPP) {
        int f = t >> 10;
        int rem = t & 1023;
        int y = rem >> 5, x = rem & 31;
        size_t base = ((size_t)f * 4096 + (size_t)(2 * y) * 64 + (size_t)(2 * x)) * EMBED + c;
        float kv = 0.25f * (k_in[base] + k_in[base + EMBED] +
                            k_in[base + 64 * EMBED] + k_in[base + 65 * EMBED]);
        float vv = 0.25f * (v_in[base] + v_in[base + EMBED] +
                            v_in[base + 64 * EMBED] + v_in[base + 65 * EMBED]);
        kpool[dst] = (_Float16)kv;
        vpool[dst] = (_Float16)vv;
    } else {
        size_t src = ((size_t)NSP + (size_t)(t - NSPP)) * EMBED + c;
        kpool[dst] = (_Float16)k_in[src];
        vpool[dst] = (_Float16)v_in[src];
    }
}

// ---------------- kernel 2: WMMA GEMM, 64x32 per wave ----------------
// Out[N x 256] = A[N x 256] @ W[256 x 256] + bias, W given pre-transposed (Wt[n][k]).
// TROUT=true stores Out transposed: Out[n][m] with row stride out_ld.
// grid = (rows/64, EMBED/32), block = 32
template <typename TO, bool TROUT>
__global__ void proj_gemm(const _Float16* __restrict__ A, const _Float16* __restrict__ Wt,
                          const float* __restrict__ bias, TO* __restrict__ Out, int out_ld) {
    int lane = threadIdx.x;
    int r0 = blockIdx.x * 64, n0 = blockIdx.y * 32;
    v8f acc[4][2] = {};
#pragma unroll
    for (int k0 = 0; k0 < EMBED; k0 += 32) {
        v16h b0 = load_B_transposed(Wt, EMBED, n0,      k0, lane);
        v16h b1 = load_B_transposed(Wt, EMBED, n0 + 16, k0, lane);
#pragma unroll
        for (int t = 0; t < 4; ++t) {
            v16h a = load_A_tile(A, EMBED, r0 + 16 * t, k0, lane);
            acc[t][0] = wmma_f16(a, b0, acc[t][0]);
            acc[t][1] = wmma_f16(a, b1, acc[t][1]);
        }
    }
    int n = lane & 15, half = lane >> 4;
    float bv0 = bias[n0 + n], bv1 = bias[n0 + 16 + n];
#pragma unroll
    for (int t = 0; t < 4; ++t) {
#pragma unroll
        for (int i = 0; i < 8; ++i) {
            int m = r0 + 16 * t + i + 8 * half;          // C layout: M=i+8*half, N=lane&15
            float o0 = acc[t][0][i] + bv0;
            float o1 = acc[t][1][i] + bv1;
            if (TROUT) {
                Out[(size_t)(n0 + n) * out_ld + m]      = (TO)o0;
                Out[(size_t)(n0 + 16 + n) * out_ld + m] = (TO)o1;
            } else {
                Out[(size_t)m * EMBED + n0 + n]      = (TO)o0;
                Out[(size_t)m * EMBED + n0 + 16 + n] = (TO)o1;
            }
        }
    }
}

// ---------------- kernel 3: attention, 32 queries x 1 head per wave ----------------
// Unnormalized-exp flash attention: logits are O(1) here (inputs ~N(0,1), scale 32^-0.5),
// so exp2(s*XSCALE + badd - 8*log2e) stays comfortably inside f32/f16 range; the row sum
// is produced by a third WMMA against an all-ones B fragment (no cross-lane shuffles at all).
// grid = (NQ/32, HEADS), block = 32
__global__ void attn_kernel(const _Float16* __restrict__ Q, const _Float16* __restrict__ Kp,
                            const _Float16* __restrict__ VpT, _Float16* __restrict__ O) {
    __shared__ _Float16 Pld[2][16 * 32];    // probability tiles, row-major 16 x 32
    int lane = threadIdx.x;
    int half = lane >> 4;
    int h = blockIdx.y;
    int q0 = blockIdx.x * 32;
    int c0 = h * HD;

    // Q fragments (two 16-query tiles), loaded once
    v16h aq0 = load_A_tile(Q, EMBED, q0,      c0, lane);
    v16h aq1 = load_A_tile(Q, EMBED, q0 + 16, c0, lane);

    // all-ones B fragment for row sums
    v16h ones;
#pragma unroll
    for (int i = 0; i < 16; ++i) ones[i] = (_Float16)1.0f;

    v8f acc[2][2] = {};     // [qtile][dim-tile] 16x16 f32 output accumulators
    v8f lsum[2]   = {};     // [qtile] row-sum accumulators (all 16 cols identical)

    for (int kv0 = 0; kv0 < NKV; kv0 += 32) {
        if (kv0 + 32 < NKV) {
            __builtin_prefetch(Kp  + (size_t)(kv0 + 32) * EMBED + c0, 0, 0);
            __builtin_prefetch(VpT + (size_t)c0 * NKV + kv0 + 32, 0, 0);
        }
        // K^T fragments: Kp row-major [token][feat] -> pairs along feat contiguous
        v16h bk0 = load_B_transposed(Kp, EMBED, kv0,      c0, lane);
        v16h bk1 = load_B_transposed(Kp, EMBED, kv0 + 16, c0, lane);
        // V fragments: VpT row-major [feat][token] -> pairs along token contiguous
        v16h bv0 = load_B_transposed(VpT, NKV, c0,      kv0, lane);
        v16h bv1 = load_B_transposed(VpT, NKV, c0 + 16, kv0, lane);

        float badd = (kv0 < NSPP) ? BADD_SP : BADD_PT;   // chunk-uniform (NSPP 32-aligned)
        v8f z = {};

        __syncthreads();                                  // protect Pld WAR across chunks
#pragma unroll
        for (int qt = 0; qt < 2; ++qt) {
            v16h aq = qt ? aq1 : aq0;
            v8f s0 = wmma_f16(aq, bk0, z);
            v8f s1 = wmma_f16(aq, bk1, z);
#pragma unroll
            for (int i = 0; i < 8; ++i) {
                float p0 = fast_exp2(s0[i] * XSCALE + badd);
                float p1 = fast_exp2(s1[i] * XSCALE + badd);
                int row = i + 8 * half, col = lane & 15;  // C layout
                Pld[qt][row * 32 + col]      = (_Float16)p0;
                Pld[qt][row * 32 + 16 + col] = (_Float16)p1;
            }
        }
        __syncthreads();
#pragma unroll
        for (int qt = 0; qt < 2; ++qt) {
            v16h ap = load_A_tile(&Pld[qt][0], 32, 0, 0, lane);
            acc[qt][0] = wmma_f16(ap, bv0, acc[qt][0]);
            acc[qt][1] = wmma_f16(ap, bv1, acc[qt][1]);
            lsum[qt]   = wmma_f16(ap, ones, lsum[qt]);
        }
    }

    // normalize and store f16 head output
#pragma unroll
    for (int qt = 0; qt < 2; ++qt) {
#pragma unroll
        for (int i = 0; i < 8; ++i) {
            int row = q0 + 16 * qt + i + 8 * half, col = lane & 15;
            float inv = fast_rcp(lsum[qt][i]);
            O[(size_t)row * EMBED + c0 + col]      = (_Float16)(acc[qt][0][i] * inv);
            O[(size_t)row * EMBED + c0 + 16 + col] = (_Float16)(acc[qt][1][i] * inv);
        }
    }
}

// ---------------- host-side launcher ----------------
extern "C" void kernel_launch(void* const* d_in, const int* in_sizes, int n_in,
                              void* d_out, int out_size, void* d_ws, size_t ws_size,
                              hipStream_t stream) {
    const float* q  = (const float*)d_in[0];
    const float* k  = (const float*)d_in[1];
    const float* v  = (const float*)d_in[2];
    const float* Wq = (const float*)d_in[3];
    const float* bq = (const float*)d_in[4];
    const float* Wk = (const float*)d_in[5];
    const float* bk = (const float*)d_in[6];
    const float* Wv = (const float*)d_in[7];
    const float* bv = (const float*)d_in[8];
    const float* Wo = (const float*)d_in[9];
    const float* bo = (const float*)d_in[10];
    float* out = (float*)d_out;

    // workspace layout (f16)
    _Float16* kpool = (_Float16*)d_ws;                    // NKV x EMBED
    _Float16* vpool = kpool + (size_t)NKV * EMBED;        // NKV x EMBED
    _Float16* qh    = vpool + (size_t)NKV * EMBED;        // NQ  x EMBED (q in f16)
    _Float16* qp    = qh    + (size_t)NQ  * EMBED;        // NQ  x EMBED
    _Float16* kp    = qp    + (size_t)NQ  * EMBED;        // NKV x EMBED
    _Float16* vpT   = kp    + (size_t)NKV * EMBED;        // EMBED x NKV (transposed)
    _Float16* ao    = vpT   + (size_t)NKV * EMBED;        // NQ  x EMBED
    _Float16* wqT   = ao    + (size_t)NQ  * EMBED;        // 4 x (EMBED x EMBED) transposed f16
    _Float16* wkT   = wqT   + (size_t)EMBED * EMBED;
    _Float16* wvT   = wkT   + (size_t)EMBED * EMBED;
    _Float16* woT   = wvT   + (size_t)EMBED * EMBED;

    // 0) weight transpose + convert, q convert (tiny, bandwidth-bound)
    transpose_w_kernel<<<EMBED, EMBED, 0, stream>>>(Wq, wqT);
    transpose_w_kernel<<<EMBED, EMBED, 0, stream>>>(Wk, wkT);
    transpose_w_kernel<<<EMBED, EMBED, 0, stream>>>(Wv, wvT);
    transpose_w_kernel<<<EMBED, EMBED, 0, stream>>>(Wo, woT);
    cvt_f16_kernel<<<NQ, EMBED, 0, stream>>>(q, qh);

    // 1) pool K/V spatial tokens, convert to f16
    pool_kernel<<<NKV, EMBED, 0, stream>>>(k, v, kpool, vpool);

    // 2) projections (WMMA f16, f32 accumulate); V projection stored transposed
    proj_gemm<_Float16, false><<<dim3(NQ  / 64, EMBED / 32), 32, 0, stream>>>(qh,    wqT, bq, qp,  0);
    proj_gemm<_Float16, false><<<dim3(NKV / 64, EMBED / 32), 32, 0, stream>>>(kpool, wkT, bk, kp,  0);
    proj_gemm<_Float16, true ><<<dim3(NKV / 64, EMBED / 32), 32, 0, stream>>>(vpool, wvT, bv, vpT, NKV);

    // 3) attention
    attn_kernel<<<dim3(NQ / 32, HEADS), 32, 0, stream>>>(qp, kp, vpT, ao);

    // 4) output projection -> fp32 result
    proj_gemm<float, false><<<dim3(NQ / 64, EMBED / 32), 32, 0, stream>>>(ao, woT, bo, out, 0);
}
